// RaftConsensusAttentionV2_34952443855215
// MI455X (gfx1250) — matrix operand order repaired and checked by reading, not compiled, not run on hardware
//
#include <hip/hip_runtime.h>
#include <cmath>

typedef __bf16 bf16_t;
typedef __attribute__((ext_vector_type(16))) __bf16 v16bf;
typedef __attribute__((ext_vector_type(8)))  __bf16 v8bf;
typedef __attribute__((ext_vector_type(8)))  float  v8f;
typedef __attribute__((ext_vector_type(4)))  float  v4f;
typedef int v4i32 __attribute__((vector_size(16)));

#define TILE  128
#define KC    32
#define LDSTR 40   // KC + 8 pad bf16 elems -> 80B row stride (16B aligned, bank-spread)

#if __has_builtin(__builtin_amdgcn_global_load_async_to_lds_b128)
#define HAVE_ASYNC_LDS 1
#else
#define HAVE_ASYNC_LDS 0
#endif

__device__ __forceinline__ void wait_async0() {
#if __has_builtin(__builtin_amdgcn_s_wait_asynccnt)
  __builtin_amdgcn_s_wait_asynccnt(0);
#else
  asm volatile("s_wait_asynccnt 0x0" ::: "memory");
#endif
}

enum { EPI_BIAS = 0, EPI_BIAS_MASK = 1, EPI_BIAS_GELU = 2, EPI_BIAS_RESID = 3 };

// C = A[M,K] @ B[K,N] + bias, epilogue variants. A row stride = lda, out row stride = ldc.
// Double-buffered LDS, software-pipelined staging: chunk i+1 global loads are issued
// before the WMMAs of chunk i and committed to LDS after them.
template <bool A_IS_F32, bool OUT_F32, int EPI>
__global__ __launch_bounds__(256) void wmma_gemm_kernel(
    const void* __restrict__ Aptr, const float* __restrict__ Bw,
    const float* __restrict__ bias, void* __restrict__ Out,
    int M, int N, int K, int lda, int ldc,
    const int* __restrict__ mask, const float* __restrict__ resid)
{
  __shared__ __align__(16) bf16_t lA[2][TILE * LDSTR];   // [row][k]
  __shared__ __align__(16) bf16_t lB[2][TILE * LDSTR];   // [col][k] (transposed)

  const int t    = threadIdx.x;
  const int lane = t & 31;
  const int wave = t >> 5;
  const int wm   = wave >> 2;   // 0..1 : 64-row slab
  const int wn   = wave & 3;    // 0..3 : 32-col slab
  const int g    = lane >> 4;   // lane half
  const int lr   = lane & 15;

  const int rowBase = blockIdx.y * TILE;
  const int colBase = blockIdx.x * TILE;

  v8f acc[4][2];
  #pragma unroll
  for (int i = 0; i < 4; ++i)
    #pragma unroll
    for (int j = 0; j < 2; ++j)
      #pragma unroll
      for (int e = 0; e < 8; ++e) acc[i][j][e] = 0.0f;

  // staging registers (chunk in flight)
  v4f  aRegF[4];
  v8bf aRegH[2];
  v4f  bReg0[2], bReg1[2];

  // ---- A stage: issue global loads (or async-to-LDS) ----
  auto issueA = [&](int k0, int buf) {
    if constexpr (A_IS_F32) {
      const float* A = (const float*)Aptr;
      #pragma unroll
      for (int i = 0; i < 4; ++i) {
        int c = t + 256 * i;  int r = c >> 3;  int kk = (c & 7) << 2;
        aRegF[i] = *(const v4f*)(A + (size_t)(rowBase + r) * lda + k0 + kk);
      }
    } else {
      const bf16_t* A = (const bf16_t*)Aptr;
      #pragma unroll
      for (int i = 0; i < 2; ++i) {
        int c = t + 256 * i;  int r = c >> 2;  int kk = (c & 3) << 3;
        const bf16_t* gp = A + (size_t)(rowBase + r) * lda + k0 + kk;
        bf16_t* lp = &lA[buf][r * LDSTR + kk];
#if HAVE_ASYNC_LDS
        __builtin_amdgcn_global_load_async_to_lds_b128(
            (__attribute__((address_space(1))) v4i32*)gp,
            (__attribute__((address_space(3))) v4i32*)lp, 0, 0);
#else
        aRegH[i] = *(const v8bf*)gp;
        (void)lp;
#endif
      }
    }
  };
  // ---- A stage: commit registers to LDS (no-op for async path) ----
  auto commitA = [&](int buf) {
    if constexpr (A_IS_F32) {
      #pragma unroll
      for (int i = 0; i < 4; ++i) {
        int c = t + 256 * i;  int r = c >> 3;  int kk = (c & 7) << 2;
        union { bf16_t h[4]; unsigned long long u; } p;
        p.h[0] = (bf16_t)aRegF[i][0]; p.h[1] = (bf16_t)aRegF[i][1];
        p.h[2] = (bf16_t)aRegF[i][2]; p.h[3] = (bf16_t)aRegF[i][3];
        *(unsigned long long*)&lA[buf][r * LDSTR + kk] = p.u;
      }
    } else {
#if !HAVE_ASYNC_LDS
      #pragma unroll
      for (int i = 0; i < 2; ++i) {
        int c = t + 256 * i;  int r = c >> 2;  int kk = (c & 3) << 3;
        *(v8bf*)&lA[buf][r * LDSTR + kk] = aRegH[i];
      }
#else
      (void)buf;
#endif
    }
  };
  // ---- B stage: issue two adjacent k-rows per thread ----
  auto issueB = [&](int k0) {
    #pragma unroll
    for (int i = 0; i < 2; ++i) {
      int c = t + 256 * i;  int k = (c >> 5) << 1;  int n4 = (c & 31) << 2;
      bReg0[i] = *(const v4f*)(Bw + (size_t)(k0 + k)     * N + colBase + n4);
      bReg1[i] = *(const v4f*)(Bw + (size_t)(k0 + k + 1) * N + colBase + n4);
    }
  };
  // ---- B stage: commit packed bf16 pairs (b32 stores, transposed) ----
  auto commitB = [&](int buf) {
    #pragma unroll
    for (int i = 0; i < 2; ++i) {
      int c = t + 256 * i;  int k = (c >> 5) << 1;  int n4 = (c & 31) << 2;
      #pragma unroll
      for (int j = 0; j < 4; ++j) {
        union { bf16_t h[2]; unsigned u; } p;
        p.h[0] = (bf16_t)bReg0[i][j];
        p.h[1] = (bf16_t)bReg1[i][j];
        *(unsigned*)&lB[buf][(n4 + j) * LDSTR + k] = p.u;
      }
    }
  };

  const int nChunks = K / KC;

  // prologue: stage chunk 0 into buffer 0
  issueA(0, 0);
  issueB(0);
  commitA(0);
  if constexpr (!A_IS_F32) { if (HAVE_ASYNC_LDS) wait_async0(); }
  commitB(0);
  __syncthreads();

  for (int ci = 0; ci < nChunks; ++ci) {
    const int buf = ci & 1;
    const int nb  = buf ^ 1;
    const bool more = (ci + 1) < nChunks;
    if (more) {                      // issue next chunk; latency hides under WMMAs
      issueA((ci + 1) * KC, nb);
      issueB((ci + 1) * KC);
    }

    // ---- fragments + WMMA for chunk ci ----
    // B 32x16 bf16: lane<16 -> col=lr, K=0..15; lane>=16 -> K=16..31
    v16bf bfrag[2];
    #pragma unroll
    for (int ns = 0; ns < 2; ++ns) {
      int bcol = wn * 32 + ns * 16 + lr;
      v8bf b0 = *(const v8bf*)&lB[buf][bcol * LDSTR + 16 * g];
      v8bf b1 = *(const v8bf*)&lB[buf][bcol * LDSTR + 16 * g + 8];
      #pragma unroll
      for (int e = 0; e < 8; ++e) { bfrag[ns][e] = b0[e]; bfrag[ns][e + 8] = b1[e]; }
    }
    // A 16x32 bf16: lane<16 -> row=lr, K=0..15; lane>=16 -> K=16..31
    #pragma unroll
    for (int ms = 0; ms < 4; ++ms) {
      int arow = wm * 64 + ms * 16 + lr;
      v8bf a0 = *(const v8bf*)&lA[buf][arow * LDSTR + 16 * g];
      v8bf a1 = *(const v8bf*)&lA[buf][arow * LDSTR + 16 * g + 8];
      v16bf afrag;
      #pragma unroll
      for (int e = 0; e < 8; ++e) { afrag[e] = a0[e]; afrag[e + 8] = a1[e]; }
      #pragma unroll
      for (int ns = 0; ns < 2; ++ns) {
        acc[ms][ns] = __builtin_amdgcn_wmma_f32_16x16x32_bf16(
            false, afrag, false, bfrag[ns], (short)0, acc[ms][ns], false, false);
      }
    }

    if (more) {                      // commit next chunk after compute
      commitA(nb);
      if constexpr (!A_IS_F32) { if (HAVE_ASYNC_LDS) wait_async0(); }
      commitB(nb);
    }
    __syncthreads();
  }

  // ---- epilogue: C/D layout: VGPR r -> row (8*g + r), lane lr -> col ----
  #pragma unroll
  for (int ms = 0; ms < 4; ++ms) {
    #pragma unroll
    for (int ns = 0; ns < 2; ++ns) {
      int col = colBase + wn * 32 + ns * 16 + lr;
      float bc = bias[col];
      #pragma unroll
      for (int r = 0; r < 8; ++r) {
        int row = rowBase + wm * 64 + ms * 16 + 8 * g + r;
        float x = acc[ms][ns][r] + bc;
        if (EPI == EPI_BIAS_MASK)  x *= (float)mask[row];
        if (EPI == EPI_BIAS_GELU)  x = 0.5f * x * (1.0f + erff(x * 0.70710678118654752f));
        if (EPI == EPI_BIAS_RESID) x += resid[(size_t)row * ldc + col];
        if (OUT_F32) ((float*)Out)[(size_t)row * ldc + col] = x;
        else         ((bf16_t*)Out)[(size_t)row * ldc + col] = (bf16_t)x;
      }
    }
  }
}

// Tiny 4-token attention: one thread per (n, query i, head h). hd = 256, H = 4.
__global__ __launch_bounds__(256) void attn4_kernel(
    const bf16_t* __restrict__ Q, const bf16_t* __restrict__ Kt,
    const bf16_t* __restrict__ V, bf16_t* __restrict__ CTX)
{
  int t = blockIdx.x * 256 + threadIdx.x;   // N*16 threads
  int n = t >> 4;
  int i = (t >> 2) & 3;
  int h = t & 3;
  const size_t base = (size_t)n * 4096 + (size_t)h * 256;  // rows are (n*4 + j), ld=1024
  const bf16_t* q = Q + base + (size_t)i * 1024;

  float s[4];
  #pragma unroll
  for (int j = 0; j < 4; ++j) {
    const bf16_t* kp = Kt + base + (size_t)j * 1024;
    float a = 0.0f;
    for (int d = 0; d < 256; d += 8) {
      v8bf qv = *(const v8bf*)(q + d);
      v8bf kv = *(const v8bf*)(kp + d);
      #pragma unroll
      for (int e = 0; e < 8; ++e) a += (float)qv[e] * (float)kv[e];
    }
    s[j] = a * 0.0625f;   // 1/sqrt(256)
  }
  float mx = fmaxf(fmaxf(s[0], s[1]), fmaxf(s[2], s[3]));
  float e0 = expf(s[0] - mx), e1 = expf(s[1] - mx);
  float e2 = expf(s[2] - mx), e3 = expf(s[3] - mx);
  float inv = 1.0f / (e0 + e1 + e2 + e3);
  float a0 = e0 * inv, a1 = e1 * inv, a2 = e2 * inv, a3 = e3 * inv;

  bf16_t* out = CTX + base + (size_t)i * 1024;
  for (int d = 0; d < 256; d += 8) {
    v8bf v0 = *(const v8bf*)(V + base + 0 * 1024 + d);
    v8bf v1 = *(const v8bf*)(V + base + 1 * 1024 + d);
    v8bf v2 = *(const v8bf*)(V + base + 2 * 1024 + d);
    v8bf v3 = *(const v8bf*)(V + base + 3 * 1024 + d);
    #pragma unroll
    for (int e = 0; e < 8; ++e)
      out[d + e] = (bf16_t)(a0 * (float)v0[e] + a1 * (float)v1[e] +
                            a2 * (float)v2[e] + a3 * (float)v3[e]);
  }
}

// In-place LayerNorm over rows of 1024 f32; one 256-thread block per row.
__global__ __launch_bounds__(256) void layernorm_kernel(
    float* __restrict__ Y, const float* __restrict__ gw, const float* __restrict__ bw)
{
  __shared__ float r1[256], r2[256];
  int row = blockIdx.x, t = threadIdx.x;
  float* y = Y + (size_t)row * 1024;
  v4f v = *(const v4f*)(y + t * 4);
  r1[t] = v.x + v.y + v.z + v.w;
  r2[t] = v.x * v.x + v.y * v.y + v.z * v.z + v.w * v.w;
  __syncthreads();
  #pragma unroll
  for (int o = 128; o > 0; o >>= 1) {
    if (t < o) { r1[t] += r1[t + o]; r2[t] += r2[t + o]; }
    __syncthreads();
  }
  float mu  = r1[0] * (1.0f / 1024.0f);
  float var = r2[0] * (1.0f / 1024.0f) - mu * mu;
  float inv = rsqrtf(var + 1e-5f);
  v4f gv = *(const v4f*)(gw + t * 4);
  v4f bv = *(const v4f*)(bw + t * 4);
  v.x = (v.x - mu) * inv * gv.x + bv.x;
  v.y = (v.y - mu) * inv * gv.y + bv.y;
  v.z = (v.z - mu) * inv * gv.z + bv.z;
  v.w = (v.w - mu) * inv * gv.w + bv.w;
  *(v4f*)(y + t * 4) = v;
}

extern "C" void kernel_launch(void* const* d_in, const int* in_sizes, int n_in,
                              void* d_out, int out_size, void* d_ws, size_t ws_size,
                              hipStream_t stream) {
  const float* features = (const float*)d_in[0];
  const int*   mask4[4] = {(const int*)d_in[1], (const int*)d_in[2],
                           (const int*)d_in[3], (const int*)d_in[4]};
  const float* Wm[4] = {(const float*)d_in[5], (const float*)d_in[7],
                        (const float*)d_in[9], (const float*)d_in[11]};
  const float* bm[4] = {(const float*)d_in[6], (const float*)d_in[8],
                        (const float*)d_in[10], (const float*)d_in[12]};
  const float* Wq = (const float*)d_in[13]; const float* bq = (const float*)d_in[14];
  const float* Wk = (const float*)d_in[15]; const float* bk = (const float*)d_in[16];
  const float* Wv = (const float*)d_in[17]; const float* bv = (const float*)d_in[18];
  const float* Wo = (const float*)d_in[19]; const float* bo = (const float*)d_in[20];
  const float* Wg1 = (const float*)d_in[21]; const float* bg1 = (const float*)d_in[22];
  const float* Wg2 = (const float*)d_in[23]; const float* bg2 = (const float*)d_in[24];
  const float* ln_g = (const float*)d_in[25]; const float* ln_b = (const float*)d_in[26];

  const int N_TOK = 8192, D = 1024;
  const size_t SZ = (size_t)N_TOK * 4 * D * sizeof(bf16_t);  // 64 MiB per [N*4, D] bf16
  if (ws_size < 4 * SZ) return;
  char* ws = (char*)d_ws;
  bf16_t* X   = (bf16_t*)(ws + 0 * SZ);
  bf16_t* Qb  = (bf16_t*)(ws + 1 * SZ);
  bf16_t* Kb  = (bf16_t*)(ws + 2 * SZ);
  bf16_t* Vb  = (bf16_t*)(ws + 3 * SZ);
  bf16_t* CTX = X;    // X dead after QKV
  bf16_t* ATT = Qb;   // Q dead after attention
  bf16_t* Hb  = Kb;   // K dead after attention
  float*  Y   = (float*)d_out;

  dim3 blk(256);

  // 1) masked modality projections -> X [N,4,D] bf16 (row stride 4D, modality offset m*D)
  {
    dim3 grd(D / TILE, N_TOK / TILE);
    for (int m = 0; m < 4; ++m)
      wmma_gemm_kernel<true, false, EPI_BIAS_MASK><<<grd, blk, 0, stream>>>(
          features, Wm[m], bm[m], X + m * D, N_TOK, D, D, D, 4 * D, mask4[m], nullptr);
  }
  // 2) QKV projections over [N*4, D]
  {
    dim3 grd(D / TILE, (N_TOK * 4) / TILE);
    wmma_gemm_kernel<false, false, EPI_BIAS><<<grd, blk, 0, stream>>>(
        X, Wq, bq, Qb, N_TOK * 4, D, D, D, D, nullptr, nullptr);
    wmma_gemm_kernel<false, false, EPI_BIAS><<<grd, blk, 0, stream>>>(
        X, Wk, bk, Kb, N_TOK * 4, D, D, D, D, nullptr, nullptr);
    wmma_gemm_kernel<false, false, EPI_BIAS><<<grd, blk, 0, stream>>>(
        X, Wv, bv, Vb, N_TOK * 4, D, D, D, D, nullptr, nullptr);
  }
  // 3) tiny 4-token attention -> CTX
  attn4_kernel<<<(N_TOK * 16) / 256, blk, 0, stream>>>(Qb, Kb, Vb, CTX);
  // 4) output projection -> ATT (== attended reshaped [N, 4D])
  wmma_gemm_kernel<false, false, EPI_BIAS><<<dim3(D / TILE, (N_TOK * 4) / TILE), blk, 0, stream>>>(
      CTX, Wo, bo, ATT, N_TOK * 4, D, D, D, D, nullptr, nullptr);
  // 5) gate MLP layer 1 + exact-erf GELU: [N,4096] x [4096,2048]
  wmma_gemm_kernel<false, false, EPI_BIAS_GELU><<<dim3(2048 / TILE, N_TOK / TILE), blk, 0, stream>>>(
      ATT, Wg1, bg1, Hb, N_TOK, 2048, 4096, 4096, 2048, nullptr, nullptr);
  // 6) gate MLP layer 2 + residual(features) -> Y f32 (d_out)
  wmma_gemm_kernel<false, true, EPI_BIAS_RESID><<<dim3(D / TILE, N_TOK / TILE), blk, 0, stream>>>(
      Hb, Wg2, bg2, Y, N_TOK, D, 2048, 2048, D, nullptr, features);
  // 7) LayerNorm in place on d_out
  layernorm_kernel<<<N_TOK, blk, 0, stream>>>(Y, ln_g, ln_b);
}
